// BayesianGRU_86260123174386
// MI455X (gfx1250) — matrix operand order repaired and checked by reading, not compiled, not run on hardware
//
#include <hip/hip_runtime.h>
#include <hip/hip_bf16.h>
#include <math.h>
#include <stdint.h>

typedef __attribute__((ext_vector_type(16))) __bf16 v16bf;
typedef __attribute__((ext_vector_type(8)))  float  v8f;

#define B_   32
#define T_   1024
#define D_   512
#define H_   512
#define G3   1536
#define NWG_REC 8
#define WPITCH 520   // 512 + 8 pad, bf16 elements (matches TDM pad: 16B per 1024B)

// workspace layout (bytes), all 256B aligned
#define OFF_CNT   0
#define OFF_HBUF  256                              // 2 * 32*512 bf16 = 65536
#define OFF_BIAS  (OFF_HBUF + 2*B_*H_*2)           // 2048 f32 = 8192
#define OFF_WIH   (OFF_BIAS + 4*H_*4)              // 512*1536 bf16 (row-major [k][n])
#define OFF_WHHT  (OFF_WIH + D_*G3*2)              // 1536*512 bf16 (transposed [n][k])
#define OFF_XG    (OFF_WHHT + D_*G3*2)             // 1024*32*1536 bf16 = 96MB

#if defined(__gfx1250__) && __has_builtin(__builtin_amdgcn_tensor_load_to_lds)
#define HAVE_TDM 1
#else
#define HAVE_TDM 0
#endif

__device__ __forceinline__ float softplusf(float x) {
  return x > 20.f ? x : log1pf(expf(x));
}

__device__ __forceinline__ unsigned pack_bf2(float a, float b) {
  union { __bf16 h[2]; unsigned u; } c;
  c.h[0] = (__bf16)a; c.h[1] = (__bf16)b;
  return c.u;
}

union FragU { v16bf v; uint4 q[2]; };

__device__ __forceinline__ v16bf make_frag(const void* p0, const void* p1) {
  FragU f;
  f.q[0] = *(const uint4*)p0;
  f.q[1] = *(const uint4*)p1;
  return f.v;
}

__device__ __forceinline__ v8f wmma_bf16(v16bf a, v16bf b, v8f c) {
  return __builtin_amdgcn_wmma_f32_16x16x32_bf16(
      false, a, false, b, (short)0, c, false, false);
}

#if HAVE_TDM
typedef unsigned int u32x4 __attribute__((ext_vector_type(4)));
typedef int          i32x4 __attribute__((ext_vector_type(4)));
typedef int          i32x8 __attribute__((ext_vector_type(8)));

// 2D TDM tile load: global (row-major, stride0 elems of 2B) -> LDS, optional
// LDS padding of 16B after every 1024B (pad_interval=256 DW, pad_amount=4 DW).
__device__ __forceinline__ void tdm_load_2d_bf16(
    unsigned lds_addr, const void* gptr,
    unsigned tile_d0, unsigned tile_d1,
    unsigned tensor_d0, unsigned tensor_d1,
    unsigned stride0, bool pad)
{
  unsigned long long ga = (unsigned long long)(uintptr_t)gptr;
  u32x4 g0;
  g0[0] = 1u;                                   // count=1 (valid user descriptor)
  g0[1] = lds_addr;                             // LDS byte address
  g0[2] = (unsigned)(ga & 0xffffffffu);         // global_addr[31:0]
  g0[3] = (unsigned)((ga >> 32) & 0x01ffffffu) | (2u << 30);  // [56:32] | type=2
  unsigned w0 = (1u << 16);                     // data_size = 2 bytes
  if (pad) w0 |= (1u << 20) | (7u << 22) | (3u << 25);
  i32x8 g1;
  g1[0] = (int)w0;
  g1[1] = (int)((tensor_d0 & 0xffffu) << 16);                        // dim0[15:0]
  g1[2] = (int)(((tensor_d0 >> 16) & 0xffffu) | ((tensor_d1 & 0xffffu) << 16));
  g1[3] = (int)(((tensor_d1 >> 16) & 0xffffu) | ((tile_d0 & 0xffffu) << 16));
  g1[4] = (int)(tile_d1 & 0xffffu);             // tile_dim1; tile_dim2 = 0
  g1[5] = (int)stride0;                         // tensor_dim0_stride[31:0]
  g1[6] = 0;
  g1[7] = 0;
  i32x4 gz = {0, 0, 0, 0};
#if __has_include(<hip/amd_detail/amd_gfx1250_TDM.h>)
  i32x8 gz8 = {0, 0, 0, 0, 0, 0, 0, 0};
  __builtin_amdgcn_tensor_load_to_lds(g0, g1, gz, gz, gz8, 0);
#else
  __builtin_amdgcn_tensor_load_to_lds(g0, g1, gz, gz, 0);
#endif
}

__device__ __forceinline__ void tdm_wait0() {
#if __has_builtin(__builtin_amdgcn_s_wait_tensorcnt)
  __builtin_amdgcn_s_wait_tensorcnt(0);
#else
  asm volatile("s_wait_tensorcnt 0x0" ::: "memory");
#endif
}
#endif // HAVE_TDM

// ---------------------------------------------------------------------------
// Kernel 1: sample Bayesian weights -> bf16, sample bias -> f32, init h buffer
// w_ih stored [k][n]; w_hh stored TRANSPOSED [n][k] (n remapped: rz | n-gate)
// ---------------------------------------------------------------------------
__global__ void gru_sample_init(
    const float* __restrict__ wim, const float* __restrict__ wir,
    const float* __restrict__ whm, const float* __restrict__ whr,
    const float* __restrict__ bm,  const float* __restrict__ br,
    const float* __restrict__ ei,  const float* __restrict__ eh,
    const float* __restrict__ eb,  const float* __restrict__ h0,
    __bf16* __restrict__ wihb, __bf16* __restrict__ whhbT,
    float* __restrict__ biass, __bf16* __restrict__ hbuf0,
    unsigned* __restrict__ counter)
{
  int i = blockIdx.x * blockDim.x + threadIdx.x;
  if (i == 0) *counter = 0u;
  if (i < 4 * H_) biass[i] = bm[i] + softplusf(br[i]) * eb[i];
  if (i < B_ * H_) hbuf0[i] = (__bf16)h0[i];
  if (i < D_ * 4 * H_) {
    int c = i & (4 * H_ - 1);          // 0..2047
    int d = i >> 11;
    float wi = wim[i] + softplusf(wir[i]) * ei[i];
    float wh = whm[i] + softplusf(whr[i]) * eh[i];
    if (c < G3) wihb[(size_t)d * G3 + c] = (__bf16)wi;
    int cp = -1;
    if (c < 1024) cp = c;              // r,z recurrent
    else if (c >= 1536) cp = c - 512;  // n recurrent -> rows 1024..1535
    if (cp >= 0) whhbT[(size_t)cp * D_ + d] = (__bf16)wh;
  }
}

// ---------------------------------------------------------------------------
// Kernel 2: xg[T][B][1536] = x[B*T][512] @ wih_bf16 + bias[col]   (bf16 WMMA)
// ---------------------------------------------------------------------------
__global__ void __launch_bounds__(256) gru_xgemm(
    const float* __restrict__ x, const __bf16* __restrict__ wihb,
    const float* __restrict__ biass, __bf16* __restrict__ xg)
{
  __shared__ __bf16 ldsA[128][40];   // [m][k], pad 8
  __shared__ __bf16 ldsB[128][40];   // [n][k] (transposed), pad 8

  const int tid  = threadIdx.x;
  const int lane = tid & 31;
  const int wv   = tid >> 5;
  const int l16  = lane & 15;
  const int half = lane >> 4;
  const int Mbase = blockIdx.x * 128;
  const int Nbase = blockIdx.y * 128;
  const int mq = wv & 1;
  const int nq = wv >> 1;

  v8f acc[4][2];
  for (int i = 0; i < 4; ++i)
    for (int j = 0; j < 2; ++j)
      for (int e = 0; e < 8; ++e) acc[i][j][e] = 0.f;

  for (int kt = 0; kt < D_ / 32; ++kt) {
    const int k0 = kt * 32;
    {
      int row0 = tid >> 3;
      int kq = (tid & 7) * 4;
      for (int it = 0; it < 4; ++it) {
        int row = row0 + 32 * it;
        float4 v = *(const float4*)(x + (size_t)(Mbase + row) * D_ + k0 + kq);
        uint2 p;
        p.x = pack_bf2(v.x, v.y);
        p.y = pack_bf2(v.z, v.w);
        *(uint2*)&ldsA[row][kq] = p;
      }
    }
    {
      int krow = tid >> 4;
      int nq8 = (tid & 15) * 8;
      for (int it = 0; it < 2; ++it) {
        int k = krow + 16 * it;
        union { uint4 q; __bf16 h[8]; } u;
        u.q = *(const uint4*)(wihb + (size_t)(k0 + k) * G3 + Nbase + nq8);
        for (int e = 0; e < 8; ++e) ldsB[nq8 + e][k] = u.h[e];
      }
    }
    __syncthreads();

    v16bf a[4], bb[2];
    for (int i = 0; i < 4; ++i) {
      int ml = mq * 64 + i * 16 + l16;
      a[i] = make_frag(&ldsA[ml][16 * half], &ldsA[ml][16 * half + 8]);
    }
    for (int j = 0; j < 2; ++j) {
      int nl = nq * 32 + j * 16 + l16;
      bb[j] = make_frag(&ldsB[nl][16 * half], &ldsB[nl][16 * half + 8]);
    }
    for (int i = 0; i < 4; ++i)
      for (int j = 0; j < 2; ++j)
        acc[i][j] = wmma_bf16(a[i], bb[j], acc[i][j]);
    __syncthreads();
  }

  for (int j = 0; j < 2; ++j) {
    int col = Nbase + nq * 32 + j * 16 + l16;
    float bv = biass[col];
    for (int i = 0; i < 4; ++i) {
      for (int e = 0; e < 8; ++e) {
        int row = Mbase + mq * 64 + i * 16 + e + 8 * half;
        int b = row >> 10;
        int t = row & 1023;
        xg[((size_t)t * B_ + b) * G3 + col] = (__bf16)(acc[i][j][e] + bv);
      }
    }
  }
}

// ---------------------------------------------------------------------------
// Kernel 3: persistent recurrence, 8 WGs. w_hh slice (192KB) TDM-loaded into
// LDS once with padded pitch; h staged per step via TDM (32KB, padded pitch).
// h carried in f32 registers; grid sync via monotonic atomic counter.
// ---------------------------------------------------------------------------
__global__ void __launch_bounds__(256) gru_recurrence(
    const float* __restrict__ h0,
    const __bf16* __restrict__ whhbT, const float* __restrict__ biass,
    const __bf16* __restrict__ xg, __bf16* __restrict__ hbuf,
    float* __restrict__ out, unsigned* __restrict__ counter)
{
  extern __shared__ __bf16 smem[];
  __bf16* ldsW = smem;                         // [192][WPITCH]
  __bf16* ldsH = smem + 192 * WPITCH;          // [32][WPITCH]

  const int g    = blockIdx.x;
  const int tid  = threadIdx.x;
  const int lane = tid & 31;
  const int wv   = tid >> 5;
  const int l16  = lane & 15;
  const int half = lane >> 4;
  const int mtile = wv & 1;
  const int nc    = wv >> 1;

  // one-time: stage this WG's w_hh slice (rows g*64 | 512+g*64 | 1024+g*64)
#if HAVE_TDM
  if (wv == 0) {
    for (int c3 = 0; c3 < 3; ++c3) {
      int grow = c3 * 512 + g * 64;            // 0,512,1024 chunk bases + g*64
      tdm_load_2d_bf16((unsigned)(uintptr_t)(ldsW + (size_t)c3 * 64 * WPITCH),
                       whhbT + (size_t)grow * D_,
                       /*tile*/ 512, 64, /*tensor*/ 512, G3,
                       /*stride0*/ 512, /*pad*/ true);
    }
    tdm_wait0();
  }
#else
  for (int c = tid; c < 192 * 64; c += 256) {  // uint4 chunks (8 bf16)
    int r = c >> 6, kq = (c & 63) * 8;
    int grow = (r >> 6) * 512 + g * 64 + (r & 63);
    *(uint4*)&ldsW[(size_t)r * WPITCH + kq] =
        *(const uint4*)&whhbT[(size_t)grow * D_ + kq];
  }
#endif
  __syncthreads();

  const int n_h = g * 64 + nc * 16 + l16;
  float h_reg[8];
  for (int e = 0; e < 8; ++e) {
    int b = mtile * 16 + e + 8 * half;
    h_reg[e] = h0[b * H_ + n_h];
  }
  const float bhn = biass[3 * H_ + n_h];

  for (int t = 0; t < T_; ++t) {
    const __bf16* hb = hbuf + (size_t)(t & 1) * (B_ * H_);
    __bf16* hnb      = hbuf + (size_t)((t + 1) & 1) * (B_ * H_);

    // stage h(t) into LDS with padded pitch
#if HAVE_TDM
    if (wv == 0) {
      tdm_load_2d_bf16((unsigned)(uintptr_t)ldsH, hb,
                       512, 32, 512, 32, 512, true);
      tdm_wait0();
    }
#else
    for (int c = tid; c < 32 * 64; c += 256) {
      int r = c >> 6, kq = (c & 63) * 8;
      *(uint4*)&ldsH[(size_t)r * WPITCH + kq] =
          *(const uint4*)&hb[(size_t)r * H_ + kq];
    }
#endif
    __syncthreads();

    v8f aR, aZ, aN;
    for (int e = 0; e < 8; ++e) { aR[e] = 0.f; aZ[e] = 0.f; aN[e] = 0.f; }

    for (int kt = 0; kt < H_ / 32; ++kt) {
      int mrow = mtile * 16 + l16;
      const __bf16* hp = ldsH + (size_t)mrow * WPITCH + kt * 32;
      v16bf af = make_frag(hp + 16 * half, hp + 16 * half + 8);
      const __bf16* wr = ldsW + (size_t)(nc * 16 + l16) * WPITCH + kt * 32 + 16 * half;
      const __bf16* wz = wr + 64 * WPITCH;
      const __bf16* wn = wr + 128 * WPITCH;
      aR = wmma_bf16(af, make_frag(wr, wr + 8), aR);
      aZ = wmma_bf16(af, make_frag(wz, wz + 8), aZ);
      aN = wmma_bf16(af, make_frag(wn, wn + 8), aN);
    }

    const __bf16* xgt = xg + (size_t)t * (B_ * G3);
    const bool last = (t == T_ - 1);
    for (int e = 0; e < 8; ++e) {
      int b = mtile * 16 + e + 8 * half;
      const __bf16* xr = xgt + (size_t)b * G3;
      float R  = (float)xr[n_h]       + aR[e];
      float Z  = (float)xr[512 + n_h] + aZ[e];
      float Nn = (float)xr[1024 + n_h];
      float r = 1.f / (1.f + expf(-R));
      float z = 1.f / (1.f + expf(-Z));
      float n = tanhf(Nn + r * (aN[e] + bhn));
      float hv = (1.f - z) * n + z * h_reg[e];
      h_reg[e] = hv;
      out[((size_t)b * T_ + t) * H_ + n_h] = hv;
      hnb[b * H_ + n_h] = (__bf16)hv;
      if (last) out[(size_t)B_ * T_ * H_ + b * H_ + n_h] = hv;
    }
    if (!last) {
      int b0 = mtile * 16 + 8 * half;
      __builtin_prefetch(xgt + (size_t)B_ * G3 + (size_t)b0 * G3 + n_h, 0, 1);
    }

    // grid-wide release / arrive / acquire
    __threadfence();
    __syncthreads();
    if (tid == 0) {
      atomicAdd(counter, 1u);
      unsigned tgt = (unsigned)(NWG_REC * (t + 1));
      while (*(volatile unsigned*)counter < tgt) __builtin_amdgcn_s_sleep(2);
    }
    __syncthreads();
    __threadfence();
#if defined(__gfx1250__) && __has_builtin(__builtin_amdgcn_s_cluster_barrier)
    __builtin_amdgcn_s_cluster_barrier();  // NOP when not cluster-dispatched
#endif
  }
}

// ---------------------------------------------------------------------------
extern "C" void kernel_launch(void* const* d_in, const int* in_sizes, int n_in,
                              void* d_out, int out_size, void* d_ws, size_t ws_size,
                              hipStream_t stream) {
  const float* x    = (const float*)d_in[0];
  const float* h0   = (const float*)d_in[1];
  const float* wim  = (const float*)d_in[2];
  const float* wir  = (const float*)d_in[3];
  const float* whm  = (const float*)d_in[4];
  const float* whr  = (const float*)d_in[5];
  const float* bm   = (const float*)d_in[6];
  const float* br   = (const float*)d_in[7];
  const float* ei   = (const float*)d_in[8];
  const float* eh   = (const float*)d_in[9];
  const float* eb   = (const float*)d_in[10];

  char* ws = (char*)d_ws;
  unsigned* counter = (unsigned*)(ws + OFF_CNT);
  __bf16*   hbuf    = (__bf16*)(ws + OFF_HBUF);
  float*    biass   = (float*)(ws + OFF_BIAS);
  __bf16*   wihb    = (__bf16*)(ws + OFF_WIH);
  __bf16*   whhbT   = (__bf16*)(ws + OFF_WHHT);
  __bf16*   xg      = (__bf16*)(ws + OFF_XG);
  float*    out     = (float*)d_out;

  gru_sample_init<<<4096, 256, 0, stream>>>(wim, wir, whm, whr, bm, br,
                                            ei, eh, eb, h0,
                                            wihb, whhbT, biass, hbuf, counter);

  gru_xgemm<<<dim3(256, 12), 256, 0, stream>>>(x, wihb, biass, xg);

  const int smem = (192 + 32) * WPITCH * 2;   // 232,960 B (< 320KB/WGP)
  hipFuncSetAttribute((const void*)gru_recurrence,
                      hipFuncAttributeMaxDynamicSharedMemorySize, smem);
  gru_recurrence<<<NWG_REC, 256, smem, stream>>>(h0, whhbT, biass, xg,
                                                 hbuf, out, counter);
}
